// MSDeformableAttention_41420664603032
// MI455X (gfx1250) — compile-verified, matrix operand order checked by reading, not run on hardware
//
#include <hip/hip_runtime.h>
#include <hip/hip_bf16.h>
#include <math.h>

typedef __attribute__((ext_vector_type(16))) _Float16 v16h;
typedef __attribute__((ext_vector_type(8)))  float    v8f;

#define BS_    16
#define LQ_    1000
#define EMBED_ 256
#define NHEAD_ 8
#define HDIM_  32
#define SUMP_  16
#define LV_    13294   // 100*100 + 50*50 + 25*25 + 13*13

// ---------------------------------------------------------------------------
// Projection GEMM: out[M,N] = Q[M,256] @ W[256,N] + bias[N], f16 WMMA, f32 acc
// Block = 256 threads = 8 waves. blockIdx.x selects a group of 8 M-tiles,
// blockIdx.y selects the N-tile.
//
// The 256x16 weight tile is staged in LDS *pre-swizzled into fragment order*:
// lane L's 16-f16 B-fragment for K-step kt is the 32 contiguous bytes at
// f16 index (kt*32 + L)*16. Fragment component mapping (16-bit B layout,
// n = L&15, half = L>>4):
//   c in 0..7  -> K = kt*32 + half*8 + c
//   c in 8..15 -> K = kt*32 + half*8 + 16 + (c-8)
// so the per-step B fetch is one aligned v16h LDS read (2x ds_load_b128).
// ---------------------------------------------------------------------------
__global__ __launch_bounds__(256) void proj_gemm_wmma(
    const float* __restrict__ Q, const float* __restrict__ W,
    const float* __restrict__ bias, float* __restrict__ out, int N)
{
    __shared__ __align__(32) _Float16 lwsw[8 * 32 * 16];   // 8 K-steps x 32 lanes x 16 f16

    const int n0 = blockIdx.y * 16;

    // Stage + swizzle: 512 groups of 8 f16; group g = (kt*64 + lane*2 + cg)
    // covers components c = cg*8 .. cg*8+7 of lane `lane` at step `kt`.
    for (int g = threadIdx.x; g < 512; g += 256) {
        const int cg   = g & 1;
        const int ln   = (g >> 1) & 31;
        const int kt   = g >> 6;
        const int n    = ln & 15;
        const int half = ln >> 4;
        const int kbase = kt * 32 + half * 8 + cg * 16;
        union { _Float16 h[8]; float4 v; } u;
        #pragma unroll
        for (int j = 0; j < 8; ++j)
            u.h[j] = (_Float16)W[(kbase + j) * N + n0 + n];
        ((float4*)lwsw)[g] = u.v;
    }
    __syncthreads();

    const int lane = threadIdx.x & 31;
    const int wv   = threadIdx.x >> 5;
    const int m0   = (blockIdx.x * 8 + wv) * 16;
    const int row  = lane & 15;            // M row held by this lane (A side)
    const int half = lane >> 4;            // K sub-block held by this lane

    const float* __restrict__ qrow = Q + (size_t)(m0 + row) * EMBED_;
    const v16h*  __restrict__ lwv  = (const v16h*)lwsw;

    v8f acc = {};
    #pragma unroll
    for (int ks = 0; ks < 8; ++ks) {
        const int kb = ks * 32 + half * 8;
        v16h a;
        #pragma unroll
        for (int c = 0; c < 8; ++c) {
            // 16-bit A fragment: comps 0..7 -> K=kb..kb+7, 8..15 -> K=kb+16..kb+23
            a[c]     = (_Float16)qrow[kb + c];
            a[c + 8] = (_Float16)qrow[kb + 16 + c];
        }
        const v16h b = lwv[ks * 32 + lane];   // one 32B fragment read
        acc = __builtin_amdgcn_wmma_f32_16x16x32_f16(
            false, a, false, b, (short)0, acc, false, false);
    }

    // C/D layout: lane holds column n = lane&15; VGPR r = row m0 + r + half*8
    const int   n  = n0 + row;
    const float bv = bias[n];
    #pragma unroll
    for (int r = 0; r < 8; ++r) {
        out[(size_t)(m0 + r + half * 8) * N + n] = acc[r] + bv;
    }
}

// ---------------------------------------------------------------------------
// Fused softmax + bilinear sampling + aggregation.
// One block per (b, q); 8 waves = 8 heads; lane = channel (HEAD_DIM = 32).
// Lanes 0..15 own one sampling point each (location + softmax weight),
// broadcast to the wave with __shfl; every tap is a coalesced 128B wave load.
// ---------------------------------------------------------------------------
__global__ __launch_bounds__(256) void msda_sample(
    const float* __restrict__ off,     // (B*Lq, 8, 16, 2)  raw offsets
    const float* __restrict__ logits,  // (B*Lq, 8, 16)     attn logits
    const float* __restrict__ ref,     // (B, Lq, 1, 4)
    const float* __restrict__ value,   // (B, LV, 256)
    float* __restrict__ out)           // (B, Lq, 256)
{
    const int bq   = blockIdx.x;
    const int b    = bq / LQ_;
    const int h    = threadIdx.x >> 5;
    const int lane = threadIdx.x & 31;

    const float cx = ref[bq * 4 + 0];
    const float cy = ref[bq * 4 + 1];
    const float rw = ref[bq * 4 + 2];
    const float rh = ref[bq * 4 + 3];

    float px = 0.f, py = 0.f, aw = -1e30f;
    if (lane < SUMP_) {
        const int p = lane;
        const float ox = off[((size_t)bq * NHEAD_ + h) * 32 + p * 2 + 0];
        const float oy = off[((size_t)bq * NHEAD_ + h) * 32 + p * 2 + 1];
        aw = logits[((size_t)bq * NHEAD_ + h) * 16 + p];
        const int lvl = p >> 2;
        const int Wl  = (lvl == 0) ? 100 : (lvl == 1) ? 50 : (lvl == 2) ? 25 : 13;
        // scale = num_points_scale(0.25) * OFFSET_SCALE(0.5) = 0.125
        const float lx = cx + ox * 0.125f * rw;
        const float ly = cy + oy * 0.125f * rh;
        px = lx * (float)Wl - 0.5f;
        py = ly * (float)Wl - 0.5f;   // H == W at every level here
    }

    // softmax over the 16 points (lanes 0..15 group; width-16 xor butterflies)
    float mx = aw;
    #pragma unroll
    for (int m = 8; m; m >>= 1) mx = fmaxf(mx, __shfl_xor(mx, m, 16));
    float e = __expf(aw - mx);
    if (lane >= SUMP_) e = 0.f;
    float s = e;
    #pragma unroll
    for (int m = 8; m; m >>= 1) s += __shfl_xor(s, m, 16);
    aw = e / s;

    const int ch  = h * HDIM_ + lane;     // value/output channel
    float acc = 0.f;

    #pragma unroll
    for (int p = 0; p < SUMP_; ++p) {
        const float pxp = __shfl(px, p, 32);
        const float pyp = __shfl(py, p, 32);
        const float wgt = __shfl(aw, p, 32);

        const int lvl    = p >> 2;
        const int Wl     = (lvl == 0) ? 100 : (lvl == 1) ? 50 : (lvl == 2) ? 25 : 13;
        const int lstart = (lvl == 0) ? 0 : (lvl == 1) ? 10000 : (lvl == 2) ? 12500 : 13125;
        const int base   = (b * LV_ + lstart) * EMBED_ + ch;

        const float xf = floorf(pxp), yf = floorf(pyp);
        const int   x0 = (int)xf,     y0 = (int)yf;
        const float fx = pxp - xf,    fy = pyp - yf;
        const float tw[4] = { (1.f - fx) * (1.f - fy), fx * (1.f - fy),
                              (1.f - fx) * fy,         fx * fy };
        #pragma unroll
        for (int t = 0; t < 4; ++t) {
            const int xi = x0 + (t & 1);
            const int yi = y0 + (t >> 1);
            const bool valid = (xi >= 0) & (xi < Wl) & (yi >= 0) & (yi < Wl);
            const int xc = xi < 0 ? 0 : (xi > Wl - 1 ? Wl - 1 : xi);
            const int yc = yi < 0 ? 0 : (yi > Wl - 1 ? Wl - 1 : yi);
            const float v = value[base + (yc * Wl + xc) * EMBED_];
            acc += valid ? wgt * tw[t] * v : 0.f;
        }
    }

    out[(size_t)bq * EMBED_ + ch] = acc;
}

// ---------------------------------------------------------------------------
extern "C" void kernel_launch(void* const* d_in, const int* in_sizes, int n_in,
                              void* d_out, int out_size, void* d_ws, size_t ws_size,
                              hipStream_t stream) {
    (void)in_sizes; (void)n_in; (void)out_size; (void)ws_size;
    const float* query  = (const float*)d_in[0];
    const float* refp   = (const float*)d_in[1];
    const float* value  = (const float*)d_in[2];
    // d_in[3] = value_spatial_shapes (int64) — compile-time constants here
    const float* W_off  = (const float*)d_in[4];
    const float* b_off  = (const float*)d_in[5];
    const float* W_attn = (const float*)d_in[6];
    const float* b_attn = (const float*)d_in[7];
    float* out = (float*)d_out;

    float* off_buf  = (float*)d_ws;                       // 16000*256 f32
    float* attn_buf = off_buf + (size_t)BS_ * LQ_ * 256;  // 16000*128 f32

    const int M = BS_ * LQ_;                // 16000, divisible by 128
    dim3 g_off (M / (16 * 8), 256 / 16);    // (125, 16)
    dim3 g_attn(M / (16 * 8), 128 / 16);    // (125, 8)
    proj_gemm_wmma<<<g_off,  256, 0, stream>>>(query, W_off,  b_off,  off_buf,  256);
    proj_gemm_wmma<<<g_attn, 256, 0, stream>>>(query, W_attn, b_attn, attn_buf, 128);

    msda_sample<<<M, 256, 0, stream>>>(off_buf, attn_buf, refp, value, out);
}
